// TopDownInferenceLayer_56203942035651
// MI455X (gfx1250) — matrix operand order repaired and checked by reading, not compiled, not run on hardware
//
#include <hip/hip_runtime.h>
#include <math.h>
#include <stdint.h>

// ---------------------------------------------------------------------------
// Types for CDNA5 WMMA (wave32): v_wmma_f32_16x16x32_bf16
// ---------------------------------------------------------------------------
typedef __bf16 bfx16 __attribute__((ext_vector_type(16)));
typedef __bf16 bfx8  __attribute__((ext_vector_type(8)));
typedef float  fx8   __attribute__((ext_vector_type(8)));

#define WMMA_BF16(a, b, c) \
  __builtin_amdgcn_wmma_f32_16x16x32_bf16(false, (a), false, (b), (short)0, (c), false, false)

// CDNA5 async global->LDS copy (VGLOBAL async encoding, tracked by ASYNCcnt).
// vdst = per-lane LDS byte offset, vaddr = 64-bit global address, saddr = off.
#define ASYNC_COPY_B128(ldsoff, gaddr) \
  asm volatile("global_load_async_to_lds_b128 %0, %1, off" :: "v"(ldsoff), "v"(gaddr) : "memory")
#define WAIT_ASYNC0() asm volatile("s_wait_asynccnt 0x0" ::: "memory")

// Problem constants (match reference)
#define BB   4
#define LL   2048
#define HID  1024
#define FFD  4096
#define NH   8
#define DH   128
#define MROWS (BB * LL)   // 8192

// ---------------------------------------------------------------------------
// Fragment loaders (row-major bf16 memory -> WMMA VGPR layouts, ISA 7.12.2)
// A (16x32): lane L holds row M=L%16; half 0 -> K {0..7,16..23}, half 1 -> {8..15,24..31}
// B (32x16): lane L holds col N=L%16; half 0 -> K {0..7,16..23}, half 1 -> {8..15,24..31}
// ---------------------------------------------------------------------------
__device__ __forceinline__ bfx16 load_a_frag(const __bf16* __restrict__ base, int ld) {
  int lane = threadIdx.x & 31;
  int half = lane >> 4;
  const __bf16* p = base + (size_t)(lane & 15) * ld + 8 * half;
  bfx8 lo = *(const bfx8*)(p);
  bfx8 hi = *(const bfx8*)(p + 16);
  bfx16 r;
#pragma unroll
  for (int i = 0; i < 8; ++i) { r[i] = lo[i]; r[i + 8] = hi[i]; }
  return r;
}

__device__ __forceinline__ bfx16 load_b_frag(const __bf16* __restrict__ base, int ld) {
  int lane = threadIdx.x & 31;
  const __bf16* p = base + (size_t)(lane & 15) * ld + 8 * (lane >> 4);
  bfx8 lo = *(const bfx8*)(p);
  bfx8 hi = *(const bfx8*)(p + 16);
  bfx16 r;
#pragma unroll
  for (int i = 0; i < 8; ++i) { r[i] = lo[i]; r[i + 8] = hi[i]; }
  return r;
}

// ---------------------------------------------------------------------------
// fp32 -> bf16 conversion
// ---------------------------------------------------------------------------
__global__ __launch_bounds__(256) void k_cvt(const float* __restrict__ s,
                                             __bf16* __restrict__ d, int n) {
  int i = blockIdx.x * 256 + threadIdx.x;
  if (i < n) d[i] = (__bf16)s[i];
}

// ---------------------------------------------------------------------------
// LayerNorm: one block per row, fp32 in, bf16 normalized out
// ---------------------------------------------------------------------------
__global__ __launch_bounds__(256) void k_ln(const float* __restrict__ x,
                                            const float* __restrict__ g,
                                            const float* __restrict__ b,
                                            __bf16* __restrict__ out) {
  const int cols = HID;
  int row = blockIdx.x;
  const float* xr = x + (size_t)row * cols;
  __shared__ float red[256];
  int tid = threadIdx.x;

  float s = 0.f;
  for (int c = tid; c < cols; c += 256) s += xr[c];
  red[tid] = s; __syncthreads();
  for (int st = 128; st > 0; st >>= 1) {
    if (tid < st) red[tid] += red[tid + st];
    __syncthreads();
  }
  float mean = red[0] / cols;
  __syncthreads();

  float v = 0.f;
  for (int c = tid; c < cols; c += 256) { float d0 = xr[c] - mean; v += d0 * d0; }
  red[tid] = v; __syncthreads();
  for (int st = 128; st > 0; st >>= 1) {
    if (tid < st) red[tid] += red[tid + st];
    __syncthreads();
  }
  float rstd = rsqrtf(red[0] / cols + 1e-6f);

  __bf16* orow = out + (size_t)row * cols;
  for (int c = tid; c < cols; c += 256)
    orow[c] = (__bf16)((xr[c] - mean) * rstd * g[c] + b[c]);
}

// ---------------------------------------------------------------------------
// Tiled WMMA GEMM with async LDS staging + double buffering.
// C[M,N] = scale*(A[MxK] @ W[NxK]^T) (+bias)(relu)(+res)
// Block = 8 waves (2 Mwaves x 4 Nwaves), block tile 128x256, K-step 32.
// LDS rows padded to 80B: A-frag lanes hit distinct bank groups (20r mod 64),
// and 80B keeps every 16B async chunk 16B-aligned.
// ---------------------------------------------------------------------------
#define LDS_ROW_E   40                      // 32 data elems + 8 pad
#define LDS_ROW_B   (LDS_ROW_E * 2)         // 80 bytes
#define A_TILE_ROWS 128
#define B_TILE_ROWS 256
#define A_TILE_E    (A_TILE_ROWS * LDS_ROW_E)
#define B_TILE_E    (B_TILE_ROWS * LDS_ROW_E)
#define BUF_E       (A_TILE_E + B_TILE_E)
#define A_TILE_BYTES (A_TILE_E * 2)
#define BUF_BYTES    (BUF_E * 2)            // 30720 B; x2 buffers = 60 KB LDS

__device__ __forceinline__ void stage_tile(const __bf16* __restrict__ A,
                                           const __bf16* __restrict__ W,
                                           int m0, int n0, int k0, int K,
                                           uint32_t ldsBase, int tid) {
  // A tile: 128 rows x 64B = 512 x 16B chunks; 256 threads x 2
#pragma unroll
  for (int r = 0; r < 2; ++r) {
    int c = tid + 256 * r;
    int row = c >> 2, part = c & 3;
    uint64_t g = (uint64_t)(A + (size_t)(m0 + row) * K + k0 + part * 8);
    uint32_t l = ldsBase + (uint32_t)(row * LDS_ROW_B + part * 16);
    ASYNC_COPY_B128(l, g);
  }
  // B tile: 256 rows x 64B = 1024 x 16B chunks; 256 threads x 4
#pragma unroll
  for (int r = 0; r < 4; ++r) {
    int c = tid + 256 * r;
    int row = c >> 2, part = c & 3;
    uint64_t g = (uint64_t)(W + (size_t)(n0 + row) * K + k0 + part * 8);
    uint32_t l = ldsBase + A_TILE_BYTES + (uint32_t)(row * LDS_ROW_B + part * 16);
    ASYNC_COPY_B128(l, g);
  }
}

template <bool BIAS, bool RELU, bool RES, bool OUTBF>
__global__ __launch_bounds__(256) void k_gemm(const __bf16* __restrict__ A,
                                              const __bf16* __restrict__ W,
                                              const float* __restrict__ bias,
                                              const float* __restrict__ res,
                                              float* __restrict__ outF,
                                              __bf16* __restrict__ outB,
                                              int M, int N, int K, float scale) {
  __shared__ __bf16 smem[2 * BUF_E];
  int tid = threadIdx.x;
  int wave = tid >> 5;
  int wm = wave >> 2, wn = wave & 3;
  int m0 = blockIdx.y * 128;
  int n0 = blockIdx.x * 256;
  // low 32 bits of the generic shared pointer = LDS byte offset (flat aperture)
  uint32_t smemBase = (uint32_t)(uintptr_t)(void*)smem;

  fx8 acc[4][4] = {};
  int nk = K / 32;

  stage_tile(A, W, m0, n0, 0, K, smemBase, tid);

  for (int kt = 0; kt < nk; ++kt) {
    WAIT_ASYNC0();           // our copies for buf[kt&1] are home
    __syncthreads();         // everyone's copies landed; prior reads of other buf done
    if (kt + 1 < nk)
      stage_tile(A, W, m0, n0, (kt + 1) * 32, K,
                 smemBase + ((kt + 1) & 1) * BUF_BYTES, tid);

    const __bf16* bufA = smem + (size_t)(kt & 1) * BUF_E;
    const __bf16* bufB = bufA + A_TILE_E;

    bfx16 af[4], bf[4];
#pragma unroll
    for (int i = 0; i < 4; ++i)
      af[i] = load_a_frag(bufA + (wm * 64 + 16 * i) * LDS_ROW_E, LDS_ROW_E);
#pragma unroll
    for (int j = 0; j < 4; ++j)
      bf[j] = load_b_frag(bufB + (wn * 64 + 16 * j) * LDS_ROW_E, LDS_ROW_E);
#pragma unroll
    for (int i = 0; i < 4; ++i)
#pragma unroll
      for (int j = 0; j < 4; ++j)
        acc[i][j] = WMMA_BF16(af[i], bf[j], acc[i][j]);
  }

  int lane = tid & 31;
  int half = lane >> 4;
  int c0 = lane & 15;
  int wm0 = m0 + wm * 64, wn0 = n0 + wn * 64;
#pragma unroll
  for (int i = 0; i < 4; ++i) {
#pragma unroll
    for (int j = 0; j < 4; ++j) {
#pragma unroll
      for (int r = 0; r < 8; ++r) {
        int row = wm0 + 16 * i + r + 8 * half;
        int col = wn0 + 16 * j + c0;
        float v = acc[i][j][r] * scale;
        if (BIAS) v += bias[col];
        if (RELU) v = v > 0.f ? v : 0.f;
        if (RES)  v += res[(size_t)row * N + col];
        if (OUTBF) outB[(size_t)row * N + col] = (__bf16)v;
        else       outF[(size_t)row * N + col] = v;
      }
    }
  }
}

// ---------------------------------------------------------------------------
// V transpose: [B][L][H*DH] bf16 -> [B][H][DH][L] bf16 (contiguous keys for PV)
// ---------------------------------------------------------------------------
__global__ __launch_bounds__(256) void k_transpose_v(const __bf16* __restrict__ v,
                                                     __bf16* __restrict__ vt, int Lk) {
  __shared__ __bf16 t[32][33];
  int bh = blockIdx.z;             // b*NH + h
  int bb = bh >> 3, h = bh & 7;
  int l0 = blockIdx.x * 32, d0 = blockIdx.y * 32;
  int tx = threadIdx.x & 31, ty = threadIdx.x >> 5;  // 32 x 8
#pragma unroll
  for (int i = 0; i < 32; i += 8)
    t[ty + i][tx] = v[((size_t)(bb * Lk + l0 + ty + i)) * HID + h * DH + d0 + tx];
  __syncthreads();
#pragma unroll
  for (int i = 0; i < 32; i += 8)
    vt[((size_t)(bh * DH + d0 + ty + i)) * Lk + l0 + tx] = t[tx][ty + i];
}

// ---------------------------------------------------------------------------
// Flash attention with async-LDS K/V staging + double buffering.
// 4 waves/block, wave = 16 q-rows, k-block = 64 keys; online softmax.
// K tile: 64 keys x 128 dh (row stride 136 elems -> banks 4n mod 64, clean)
// V tile: 128 dh x 64 keys from V^T (row stride 72 elems -> banks 36n mod 64)
// B-fragments are software-pipelined (2-deep ping-pong) so the next
// ds_load_b128 pair is in flight while the current WMMA executes.
// Q pre-scaled by DH^-0.5. grid = (L/64, H, B), block = 128.
// ---------------------------------------------------------------------------
#define KT_STRIDE 136                 // 128 + 8 pad (272 B, 16B aligned)
#define VT_STRIDE 72                  // 64 + 8 pad (144 B, 16B aligned)
#define KT_E (64 * KT_STRIDE)         // 8704 elems
#define VT_E (128 * VT_STRIDE)        // 9216 elems
#define KV_BUF_E (KT_E + VT_E)        // 17920 elems
#define KT_BYTES (KT_E * 2)           // 17408 B
#define KV_BUF_BYTES (KV_BUF_E * 2)   // 35840 B; x2 buffers = 71680 B

__device__ __forceinline__ void stage_kv(const __bf16* __restrict__ Km,
                                         const __bf16* __restrict__ Vt,
                                         int b, int h, int kb, int L,
                                         uint32_t ldsBase, int tid) {
  // K tile: 64 rows x 256B = 1024 x 16B chunks; 128 threads x 8
#pragma unroll
  for (int r = 0; r < 8; ++r) {
    int c = tid + 128 * r;
    int row = c >> 4, part = c & 15;
    uint64_t g = (uint64_t)(Km + ((size_t)(b * L + kb + row)) * HID + h * DH + part * 8);
    uint32_t l = ldsBase + (uint32_t)(row * (KT_STRIDE * 2) + part * 16);
    ASYNC_COPY_B128(l, g);
  }
  // V tile: 128 rows x 128B = 1024 x 16B chunks; 128 threads x 8
#pragma unroll
  for (int r = 0; r < 8; ++r) {
    int c = tid + 128 * r;
    int row = c >> 3, part = c & 7;
    uint64_t g = (uint64_t)(Vt + ((size_t)((b * NH + h) * DH + row)) * L + kb + part * 8);
    uint32_t l = ldsBase + KT_BYTES + (uint32_t)(row * (VT_STRIDE * 2) + part * 16);
    ASYNC_COPY_B128(l, g);
  }
}

__global__ __launch_bounds__(128) void k_attn(const __bf16* __restrict__ Q,
                                              const __bf16* __restrict__ Km,
                                              const __bf16* __restrict__ Vt,
                                              const unsigned char* __restrict__ mask,
                                              __bf16* __restrict__ ctx, int L) {
  int b = blockIdx.z, h = blockIdx.y;
  int tid = threadIdx.x;
  int wave = tid >> 5;
  int lane = tid & 31, half = lane >> 4, c0 = lane & 15;
  int q0 = blockIdx.x * 64 + wave * 16;

  __shared__ __bf16 kvs[2 * KV_BUF_E];   // double-buffered K+V tiles
  __shared__ __bf16 pLds[4][16][64];     // per-wave P staging (C-layout -> A-layout)
  uint32_t kvBase = (uint32_t)(uintptr_t)(void*)kvs;

  const __bf16* Qb = Q + ((size_t)(b * L + q0)) * HID + h * DH;
  bfx16 aq[4];
#pragma unroll
  for (int kk = 0; kk < 4; ++kk) aq[kk] = load_a_frag(Qb + 32 * kk, HID);

  fx8 o[8] = {};
  float rm[8], rs[8];
#pragma unroll
  for (int r = 0; r < 8; ++r) { rm[r] = -1e30f; rs[r] = 0.f; }

  int nkb = L / 64;
  stage_kv(Km, Vt, b, h, 0, L, kvBase, tid);

  for (int kbt = 0; kbt < nkb; ++kbt) {
    int kb = kbt * 64;
    WAIT_ASYNC0();           // our copies for buf[kbt&1] are home
    __syncthreads();         // all copies landed; prior reads of other buf done
    if (kbt + 1 < nkb)
      stage_kv(Km, Vt, b, h, kb + 64, L,
               kvBase + ((kbt + 1) & 1) * KV_BUF_BYTES, tid);

    const __bf16* kTile = kvs + (size_t)(kbt & 1) * KV_BUF_E;
    const __bf16* vTile = kTile + KT_E;

    // S = Q @ K^T (16 x 64 per wave); K-frags ping-ponged from LDS
    fx8 s[4] = {};
    {
      bfx16 bk = load_b_frag(kTile, KT_STRIDE);
#pragma unroll
      for (int nt = 0; nt < 4; ++nt)
#pragma unroll
        for (int kk = 0; kk < 4; ++kk) {
          int nxt = nt * 4 + kk + 1;
          bfx16 bkn = bk;
          if (nxt < 16)
            bkn = load_b_frag(kTile + (16 * (nxt >> 2)) * KT_STRIDE + 32 * (nxt & 3),
                              KT_STRIDE);
          s[nt] = WMMA_BF16(aq[kk], bk, s[nt]);
          bk = bkn;
        }
    }
    // mask
#pragma unroll
    for (int nt = 0; nt < 4; ++nt)
#pragma unroll
      for (int r = 0; r < 8; ++r) {
        int row = q0 + r + 8 * half;
        int key = kb + 16 * nt + c0;
        if (mask[(size_t)(b * L + row) * L + key]) s[nt][r] = -1e9f;
      }
    // row max (reduce across 16-lane half: row lives in one half)
    float tm[8];
#pragma unroll
    for (int r = 0; r < 8; ++r) {
      float m = fmaxf(fmaxf(s[0][r], s[1][r]), fmaxf(s[2][r], s[3][r]));
#pragma unroll
      for (int x = 1; x < 16; x <<= 1) m = fmaxf(m, __shfl_xor(m, x, 32));
      tm[r] = m;
    }
#pragma unroll
    for (int r = 0; r < 8; ++r) {
      float nm = fmaxf(rm[r], tm[r]);
      float alpha = __expf(rm[r] - nm);
      rm[r] = nm;
      rs[r] *= alpha;
#pragma unroll
      for (int j = 0; j < 8; ++j) o[j][r] *= alpha;
    }
    // P = exp(S - rowmax); stage to LDS; accumulate row sums
    float psum[8];
#pragma unroll
    for (int r = 0; r < 8; ++r) psum[r] = 0.f;
#pragma unroll
    for (int nt = 0; nt < 4; ++nt)
#pragma unroll
      for (int r = 0; r < 8; ++r) {
        float p = __expf(s[nt][r] - rm[r]);
        psum[r] += p;
        pLds[wave][r + 8 * half][16 * nt + c0] = (__bf16)p;
      }
#pragma unroll
    for (int r = 0; r < 8; ++r) {
      float ps = psum[r];
#pragma unroll
      for (int x = 1; x < 16; x <<= 1) ps += __shfl_xor(ps, x, 32);
      rs[r] += ps;
    }
    // O += P @ V; V-frags ping-ponged from LDS (rows = dh, contiguous keys)
#pragma unroll
    for (int kk = 0; kk < 2; ++kk) {
      bfx16 pa = load_a_frag(&pLds[wave][0][0] + 32 * kk, 64);
      bfx16 bv = load_b_frag(vTile + 32 * kk, VT_STRIDE);
#pragma unroll
      for (int j = 0; j < 8; ++j) {
        bfx16 bvn = bv;
        if (j < 7)
          bvn = load_b_frag(vTile + (16 * (j + 1)) * VT_STRIDE + 32 * kk, VT_STRIDE);
        o[j] = WMMA_BF16(pa, bv, o[j]);
        bv = bvn;
      }
    }
  }
  // normalize & write context (bf16, feeds O-projection GEMM)
#pragma unroll
  for (int j = 0; j < 8; ++j)
#pragma unroll
    for (int r = 0; r < 8; ++r) {
      int row = q0 + r + 8 * half;
      float denom = rs[r];
      float v = (denom > 0.f) ? o[j][r] / denom : 0.f;
      ctx[((size_t)(b * L + row)) * HID + h * DH + 16 * j + c0] = (__bf16)v;
    }
}

// ---------------------------------------------------------------------------
// Host orchestration
// ---------------------------------------------------------------------------
extern "C" void kernel_launch(void* const* d_in, const int* in_sizes, int n_in,
                              void* d_out, int out_size, void* d_ws, size_t ws_size,
                              hipStream_t stream) {
  const float* x     = (const float*)d_in[0];
  const float* top   = (const float*)d_in[1];
  const unsigned char* imask = (const unsigned char*)d_in[2];
  const float* ln1_g = (const float*)d_in[3];
  const float* ln1_b = (const float*)d_in[4];
  const float* wq1 = (const float*)d_in[5];
  const float* wk1 = (const float*)d_in[6];
  const float* wv1 = (const float*)d_in[7];
  const float* wo1 = (const float*)d_in[8];
  const float* ln2_g = (const float*)d_in[9];
  const float* ln2_b = (const float*)d_in[10];
  const float* wq2 = (const float*)d_in[11];
  const float* wk2 = (const float*)d_in[12];
  const float* wv2 = (const float*)d_in[13];
  const float* wo2 = (const float*)d_in[14];
  const float* ln3_g = (const float*)d_in[15];
  const float* ln3_b = (const float*)d_in[16];
  const float* w1 = (const float*)d_in[17];
  const float* b1 = (const float*)d_in[18];
  const float* w2 = (const float*)d_in[19];
  const float* b2 = (const float*)d_in[20];
  float* out = (float*)d_out;

  // --- workspace layout (bytes, 256-aligned) ---
  char* ws = (char*)d_ws;
  size_t off = 0;
  auto alloc = [&](size_t bytes) { size_t o = off; off = (off + bytes + 255) & ~(size_t)255; return o; };
  const size_t HH = (size_t)HID * HID;        // 1M elems
  __bf16* wqb1 = (__bf16*)(ws + alloc(HH * 2));
  __bf16* wkb1 = (__bf16*)(ws + alloc(HH * 2));
  __bf16* wvb1 = (__bf16*)(ws + alloc(HH * 2));
  __bf16* wob1 = (__bf16*)(ws + alloc(HH * 2));
  __bf16* wqb2 = (__bf16*)(ws + alloc(HH * 2));
  __bf16* wkb2 = (__bf16*)(ws + alloc(HH * 2));
  __bf16* wvb2 = (__bf16*)(ws + alloc(HH * 2));
  __bf16* wob2 = (__bf16*)(ws + alloc(HH * 2));
  __bf16* w1b  = (__bf16*)(ws + alloc((size_t)FFD * HID * 2));
  __bf16* w2b  = (__bf16*)(ws + alloc((size_t)HID * FFD * 2));
  const size_t ACT = (size_t)MROWS * HID;     // 8M elems
  __bf16* topb = (__bf16*)(ws + alloc(ACT * 2));
  __bf16* yb   = (__bf16*)(ws + alloc(ACT * 2));
  float*  xacc = (float*)(ws + alloc(ACT * 4));
  // pool: q,k,v,vt,ctx (bf16) during attention; reused as FFN hidden h later
  char* pool = ws + alloc(5 * ACT * 2);
  __bf16* qb  = (__bf16*)(pool + 0 * ACT * 2);
  __bf16* kb  = (__bf16*)(pool + 1 * ACT * 2);
  __bf16* vb  = (__bf16*)(pool + 2 * ACT * 2);
  __bf16* vtb = (__bf16*)(pool + 3 * ACT * 2);
  __bf16* ctx = (__bf16*)(pool + 4 * ACT * 2);
  __bf16* hb  = (__bf16*)pool;                // [8192, 4096] bf16 (64MB <= 80MB pool)

  const float QSCALE = 0.08838834764831845f;  // DH^-0.5
  dim3 gGemmH(HID / 256, MROWS / 128);        // N=1024
  dim3 gGemmF(FFD / 256, MROWS / 128);        // N=4096
  dim3 gAttn(LL / 64, NH, BB);
  dim3 gTr(LL / 32, DH / 32, BB * NH);

  // 0) convert weights + top_level_rep to bf16
  struct { const float* s; __bf16* d; int n; } cv[] = {
    {wq1, wqb1, (int)HH}, {wk1, wkb1, (int)HH}, {wv1, wvb1, (int)HH}, {wo1, wob1, (int)HH},
    {wq2, wqb2, (int)HH}, {wk2, wkb2, (int)HH}, {wv2, wvb2, (int)HH}, {wo2, wob2, (int)HH},
    {w1, w1b, FFD * HID}, {w2, w2b, HID * FFD}, {top, topb, (int)ACT},
  };
  for (auto& c : cv)
    k_cvt<<<(c.n + 255) / 256, 256, 0, stream>>>(c.s, c.d, c.n);

  // ===== block 1: self-attention =====
  k_ln<<<MROWS, 256, 0, stream>>>(x, ln1_g, ln1_b, yb);
  k_gemm<false, false, false, true><<<gGemmH, 256, 0, stream>>>(yb, wqb1, nullptr, nullptr, nullptr, qb, MROWS, HID, HID, QSCALE);
  k_gemm<false, false, false, true><<<gGemmH, 256, 0, stream>>>(yb, wkb1, nullptr, nullptr, nullptr, kb, MROWS, HID, HID, 1.f);
  k_gemm<false, false, false, true><<<gGemmH, 256, 0, stream>>>(yb, wvb1, nullptr, nullptr, nullptr, vb, MROWS, HID, HID, 1.f);
  k_transpose_v<<<gTr, 256, 0, stream>>>(vb, vtb, LL);
  k_attn<<<gAttn, 128, 0, stream>>>(qb, kb, vtb, imask, ctx, LL);
  k_gemm<false, false, true, false><<<gGemmH, 256, 0, stream>>>(ctx, wob1, nullptr, x, xacc, nullptr, MROWS, HID, HID, 1.f);

  // ===== block 2: cross attention (K/V from top_level_rep) =====
  k_ln<<<MROWS, 256, 0, stream>>>(xacc, ln2_g, ln2_b, yb);
  k_gemm<false, false, false, true><<<gGemmH, 256, 0, stream>>>(yb,   wqb2, nullptr, nullptr, nullptr, qb, MROWS, HID, HID, QSCALE);
  k_gemm<false, false, false, true><<<gGemmH, 256, 0, stream>>>(topb, wkb2, nullptr, nullptr, nullptr, kb, MROWS, HID, HID, 1.f);
  k_gemm<false, false, false, true><<<gGemmH, 256, 0, stream>>>(topb, wvb2, nullptr, nullptr, nullptr, vb, MROWS, HID, HID, 1.f);
  k_transpose_v<<<gTr, 256, 0, stream>>>(vb, vtb, LL);
  k_attn<<<gAttn, 128, 0, stream>>>(qb, kb, vtb, imask, ctx, LL);
  k_gemm<false, false, true, false><<<gGemmH, 256, 0, stream>>>(ctx, wob2, nullptr, xacc, xacc, nullptr, MROWS, HID, HID, 1.f);

  // ===== block 3: FFN =====
  k_ln<<<MROWS, 256, 0, stream>>>(xacc, ln3_g, ln3_b, yb);
  k_gemm<true, true, false, true><<<gGemmF, 256, 0, stream>>>(yb, w1b, b1, nullptr, nullptr, hb, MROWS, FFD, HID, 1.f);
  k_gemm<true, false, true, false><<<gGemmH, 256, 0, stream>>>(hb, w2b, b2, xacc, out, nullptr, MROWS, HID, FFD, 1.f);
}
